// VectorQuantizer_38946763440497
// MI455X (gfx1250) — matrix operand order repaired and checked by reading, compiled, not run on hardware
//
#include <hip/hip_runtime.h>

typedef __attribute__((ext_vector_type(16))) __bf16 v16bf;
typedef __attribute__((ext_vector_type(2)))  __bf16 v2bf;
typedef __attribute__((ext_vector_type(8)))  float  v8f;

#define DDIM      128
#define KCODES    1024
#define TILE_K    16
#define NTILES    (KCODES / TILE_K)     // 64
#define WAVES     8
#define THREADS   256
#define ROWS_PER_BLOCK (WAVES * 16)     // 128
#define CBS       136                   // padded LDS row stride in bf16 elems (bank-conflict-free)

// f32 pair -> packed bf16 (RNE): let the compiler's native __bf16 fptrunc pick
// the HW convert if gfx1250 has one; otherwise it legalizes to the int sequence.
__device__ __forceinline__ unsigned int pack2(float lo, float hi) {
  union { v2bf v; unsigned int u; } cv;
  cv.v.x = (__bf16)lo;
  cv.v.y = (__bf16)hi;
  return cv.u;
}

union AFrag { v16bf v; unsigned int u[8]; };
union BFrag { v16bf v; uint4 q[2]; };

// ---------------- c_sq = ||codebook_k||^2 (exact f32) ----------------
__global__ __launch_bounds__(256) void vq_csq_kernel(const float* __restrict__ cb,
                                                     float* __restrict__ csq) {
  int k = blockIdx.x * blockDim.x + threadIdx.x;
  if (k < KCODES) {
    const float4* row = (const float4*)(cb + (size_t)k * DDIM);
    float s = 0.0f;
#pragma unroll
    for (int i = 0; i < DDIM / 4; ++i) {
      float4 v = row[i];
      s += v.x * v.x + v.y * v.y + v.z * v.z + v.w * v.w;
    }
    csq[k] = s;
  }
}

// ---------------- main: WMMA cross-term + argmin + gather ----------------
__global__ __launch_bounds__(THREADS) void vq_main_kernel(const float* __restrict__ inp,
                                                          const float* __restrict__ cb,
                                                          const float* __restrict__ csq,
                                                          float* __restrict__ out) {
  __shared__ unsigned short s_cb[2][TILE_K * CBS];  // bf16 codebook tiles (double buffered)
  __shared__ float s_csq[KCODES];

  const int tid    = threadIdx.x;
  const int wave   = tid >> 5;
  const int lane   = tid & 31;
  const int lane16 = lane & 15;
  const int half   = lane >> 4;

  // preload exact ||c||^2 table
  for (int i = tid; i < KCODES; i += THREADS) s_csq[i] = csq[i];

  const int m0 = blockIdx.x * ROWS_PER_BLOCK + wave * 16;

  // ---- A fragments: 4 K-slabs of 16x32 bf16, converted once, kept in VGPRs ----
  // Layout (ISA 7.12.2, 16-bit A 16x32): lane half h: VGPR0..3 = K 8h..8h+7,
  // VGPR4..7 = K 16+8h..16+8h+7; M = lane%16.
  AFrag afr[4];
  {
    const float* base = inp + (size_t)(m0 + lane16) * DDIM + 8 * half;
#pragma unroll
    for (int s = 0; s < 4; ++s) {
      float4 x0 = *(const float4*)(base + 32 * s);
      float4 x1 = *(const float4*)(base + 32 * s + 4);
      float4 y0 = *(const float4*)(base + 32 * s + 16);
      float4 y1 = *(const float4*)(base + 32 * s + 20);
      afr[s].u[0] = pack2(x0.x, x0.y);
      afr[s].u[1] = pack2(x0.z, x0.w);
      afr[s].u[2] = pack2(x1.x, x1.y);
      afr[s].u[3] = pack2(x1.z, x1.w);
      afr[s].u[4] = pack2(y0.x, y0.y);
      afr[s].u[5] = pack2(y0.z, y0.w);
      afr[s].u[6] = pack2(y1.x, y1.y);
      afr[s].u[7] = pack2(y1.z, y1.w);
    }
  }

  // stage tile 0 (f32 -> bf16) into LDS buffer 0
  const int scode = tid >> 4;          // 0..15 : code within tile
  const int sd0   = (tid & 15) * 8;    // dim offset, 8 floats per thread
  {
    const float4* g = (const float4*)(cb + (size_t)scode * DDIM + sd0);
    float4 a0 = g[0], a1 = g[1];
    uint4 q = { pack2(a0.x, a0.y), pack2(a0.z, a0.w),
                pack2(a1.x, a1.y), pack2(a1.z, a1.w) };
    *(uint4*)&s_cb[0][scode * CBS + sd0] = q;
  }
  __syncthreads();

  float md[8];
  int   mi[8];
#pragma unroll
  for (int v = 0; v < 8; ++v) { md[v] = 3.4e38f; mi[v] = 0; }

  for (int t = 0; t < NTILES; ++t) {
    // prefetch next codebook tile into registers (overlaps WMMA below)
    float4 p0 = {0, 0, 0, 0}, p1 = {0, 0, 0, 0};
    if (t + 1 < NTILES) {
      const float4* g =
          (const float4*)(cb + (size_t)((t + 1) * TILE_K + scode) * DDIM + sd0);
      p0 = g[0]; p1 = g[1];
    }

    // ---- 4 accumulating bf16 WMMAs over D=128 ----
    // B layout (32x16 bf16): lane N = lane%16; lane half h: K 8h..8h+7, then K 16+8h..16+8h+7
    const unsigned short* brow = &s_cb[t & 1][lane16 * CBS + 8 * half];
    v8f acc = {};
#pragma unroll
    for (int s = 0; s < 4; ++s) {
      BFrag b;
      b.q[0] = *(const uint4*)(brow + 32 * s);
      b.q[1] = *(const uint4*)(brow + 32 * s + 16);
      acc = __builtin_amdgcn_wmma_f32_16x16x32_bf16(
          false, afr[s].v, false, b.v, (short)0, acc, false, false);
    }

    // dist' = ||c||^2 - 2*<z,c>   (||z||^2 is row-constant: drops from argmin)
    const float cs  = s_csq[t * TILE_K + lane16];
    const int   idx = t * TILE_K + lane16;
#pragma unroll
    for (int v = 0; v < 8; ++v) {
      float dist = fmaf(-2.0f, acc[v], cs);
      if (dist < md[v]) { md[v] = dist; mi[v] = idx; }  // strict < => first occurrence wins
    }

    // commit prefetched tile into the other LDS buffer
    if (t + 1 < NTILES) {
      uint4 q = { pack2(p0.x, p0.y), pack2(p0.z, p0.w),
                  pack2(p1.x, p1.y), pack2(p1.z, p1.w) };
      *(uint4*)&s_cb[(t + 1) & 1][scode * CBS + sd0] = q;
    }
    __syncthreads();
  }

  // ---- argmin across the 16 lanes holding each row (ties -> lower index) ----
#pragma unroll
  for (int v = 0; v < 8; ++v) {
#pragma unroll
    for (int off = 1; off < 16; off <<= 1) {
      float od = __shfl_xor(md[v], off, 32);
      int   oi = __shfl_xor(mi[v], off, 32);
      if (od < md[v] || (od == md[v] && oi < mi[v])) { md[v] = od; mi[v] = oi; }
    }
  }
  // row M = v + 8*half's winner now replicated in every lane of that half.

  // ---- gather: each wave writes its 16 rows, 32 lanes x float4 per row ----
#pragma unroll
  for (int r = 0; r < 16; ++r) {
    const int src = (r >> 3) << 4;               // lane 0 or lane 16
    const int k   = __shfl(mi[r & 7], src, 32);  // broadcast winning code
    float4 val = *(const float4*)(cb + (size_t)k * DDIM + lane * 4);
    *(float4*)(out + (size_t)(m0 + r) * DDIM + lane * 4) = val;
  }
}

extern "C" void kernel_launch(void* const* d_in, const int* in_sizes, int n_in,
                              void* d_out, int out_size, void* d_ws, size_t ws_size,
                              hipStream_t stream) {
  const float* inp = (const float*)d_in[0];   // [32,32,32,128] f32
  const float* cb  = (const float*)d_in[1];   // [1024,128] f32
  float* out = (float*)d_out;                 // [32,32,32,128] f32
  float* csq = (float*)d_ws;                  // 1024 f32 scratch

  vq_csq_kernel<<<(KCODES + 255) / 256, 256, 0, stream>>>(cb, csq);

  const int nvec = 32 * 32 * 32;              // 32768 rows
  vq_main_kernel<<<nvec / ROWS_PER_BLOCK, THREADS, 0, stream>>>(inp, cb, csq, out);
}
